// Network_6631429505499
// MI455X (gfx1250) — matrix operand order repaired and checked by reading, not compiled, not run on hardware
//
#include <hip/hip_runtime.h>
#include <hip/hip_bf16.h>

// ---------------------------------------------------------------------------
// R-GCN forward for MI455X (gfx1250, wave32, WMMA).
// Algebraic fusions:
//   R        = rel_wt @ (emb_e @ W_e_init)            -> [200,128] table
//   layer0:  M0 = relu(emb_h @ (W_h_init@aggre_W)+b)  -> per NODE, not edge
//   layer1:  M1 = relu(node_embed @ aggre_W + b)      -> per NODE
//   msg(e)   = M[src(e)] + R[etype(e)];  segment-mean via f32 atomics
// All dense GEMMs run on v_wmma_f32_16x16x32_bf16 (fp32 accum), with 2-row-
// strip blocking so every B fragment load feeds two back-to-back WMMAs.
// ---------------------------------------------------------------------------

#define NNODES 100000
#define NEDGES 500000
#define NRELS  200
#define NBASE  50
#define INITD  100
#define FEATD  128
#define NCLS   50
#define BNEPS  1e-5f

typedef __attribute__((ext_vector_type(16))) __bf16 v16bf;
typedef __attribute__((ext_vector_type(8)))  float  v8f;

union Frag32B { v16bf v; uint4 q[2]; };

// ------------------------------ tiny helpers -------------------------------

__global__ void zero_f32(float* __restrict__ p, long n) {
  long i = (long)blockIdx.x * blockDim.x + threadIdx.x;
  long s = (long)gridDim.x * blockDim.x;
  for (; i < n; i += s) p[i] = 0.f;
}

// Naive small GEMM: C[M,N] = A[M,K] @ B[K,N]  (only used for <=200x128 shapes)
__global__ void gemm_small(const float* __restrict__ A,
                           const float* __restrict__ B,
                           float* __restrict__ C, int M, int K, int N) {
  int idx = blockIdx.x * blockDim.x + threadIdx.x;
  if (idx >= M * N) return;
  int i = idx / N, j = idx % N;
  float s = 0.f;
  for (int k = 0; k < K; ++k) s += A[i * K + k] * B[k * N + j];
  C[idx] = s;
}

// Wt[Np,Kp] (row-major, bf16, zero padded) = transpose of W[K,N] (f32)
__global__ void transpose_to_bf16(const float* __restrict__ W,
                                  __bf16* __restrict__ Wt,
                                  int K, int N, int Kp, int Np) {
  int idx = blockIdx.x * blockDim.x + threadIdx.x;
  if (idx >= Np * Kp) return;
  int n = idx / Kp, k = idx % Kp;
  float v = (n < N && k < K) ? W[k * N + n] : 0.f;
  Wt[idx] = (__bf16)v;
}

// Y[rows,Kp] bf16 (zero padded) = X[rows,K] f32
__global__ void convert_pad_bf16(const float* __restrict__ X,
                                 __bf16* __restrict__ Y,
                                 long rows, int K, int Kp) {
  long idx = (long)blockIdx.x * blockDim.x + threadIdx.x;
  if (idx >= rows * Kp) return;
  int  k = (int)(idx % Kp);
  long r = idx / Kp;
  Y[idx] = (__bf16)((k < K) ? X[r * K + k] : 0.f);
}

// ------------------------------ WMMA GEMM ----------------------------------
// One wave computes a (RT*16)-row x (NT*16)-col strip of C = A @ B.
//   A : [M, Kp]  row-major bf16 (Kp % 32 == 0, M % (RT*16) == 0 handled by
//       the row0 >= M early-out; all launches here have exact multiples)
//   Bt: [NT*16, Kp] row-major bf16 = B transposed (padded cols are zero rows)
// Epilogue: + bias, optional relu; writes f32 (ldc/ncols-clipped) and/or bf16.
//
// Fragment layouts per CDNA5 ISA 7.12.2:
//   A 16x32 bf16: lane L: m=L%16, h=L/16; VGPR0-3 hold k = kb+8h..+7,
//                 VGPR4-7 hold k = kb+16+8h..+7  -> two contiguous 16B loads.
//   B 32x16 bf16: lane L: n=L%16, h=L/16; VGPR0-3 hold k = kb+8h..+7,
//                 VGPR4-7 hold k = kb+16+8h..+7 -> two 16B loads from transposed weights.
//   C/D f32: vector elem r -> row 8h+r, col n=L%16.
//
// RT row strips share each B fragment: one B load -> RT back-to-back WMMAs,
// halving (RT=2) the load-wait frequency per WMMA regardless of how the
// scheduler serializes the B loads.
template <int NT, int RT>
__global__ __launch_bounds__(256) void wmma_gemm(
    const __bf16* __restrict__ A, int Kp,
    const __bf16* __restrict__ Bt,
    const float* __restrict__ bias,
    float* __restrict__ Cf, int ldc, int ncols,
    __bf16* __restrict__ Cb,
    int M, int relu) {
  int lane = threadIdx.x & 31;
  int wid  = blockIdx.x * (blockDim.x >> 5) + (threadIdx.x >> 5);
  int row0 = wid * (16 * RT);
  if (row0 >= M) return;
  int m = lane & 15, h = lane >> 4;

  const __bf16* arow[RT];
#pragma unroll
  for (int rt = 0; rt < RT; ++rt)
    arow[rt] = A + (size_t)(row0 + rt * 16 + m) * Kp + 8 * h;

  // Hoisted per-lane B row pointers (k-invariant part of the address math).
  const __bf16* brow[NT];
#pragma unroll
  for (int t = 0; t < NT; ++t)
    brow[t] = Bt + (size_t)(t * 16 + m) * Kp + 8 * h;

  v8f acc[RT][NT];
#pragma unroll
  for (int rt = 0; rt < RT; ++rt)
#pragma unroll
    for (int t = 0; t < NT; ++t)
      acc[rt][t] = (v8f){0.f,0.f,0.f,0.f,0.f,0.f,0.f,0.f};

  for (int kb = 0; kb < Kp; kb += 32) {
    Frag32B a[RT];
#pragma unroll
    for (int rt = 0; rt < RT; ++rt) {
      a[rt].q[0] = *(const uint4*)(arow[rt] + kb);
      a[rt].q[1] = *(const uint4*)(arow[rt] + kb + 16);
    }
#pragma unroll
    for (int t = 0; t < NT; ++t) {
      Frag32B b;
      b.q[0] = *(const uint4*)(brow[t] + kb);
      b.q[1] = *(const uint4*)(brow[t] + kb + 16);
#pragma unroll
      for (int rt = 0; rt < RT; ++rt)
        acc[rt][t] = __builtin_amdgcn_wmma_f32_16x16x32_bf16(
            false, a[rt].v, false, b.v, (short)0, acc[rt][t], false, false);
    }
  }

#pragma unroll
  for (int rt = 0; rt < RT; ++rt) {
#pragma unroll
    for (int t = 0; t < NT; ++t) {
#pragma unroll
      for (int r = 0; r < 8; ++r) {
        int row = row0 + rt * 16 + 8 * h + r;
        int col = t * 16 + m;
        if (col < ncols) {
          float v = acc[rt][t][r];
          if (bias) v += bias[col];
          if (relu) v = v > 0.f ? v : 0.f;
          if (Cf) Cf[(size_t)row * ldc + col] = v;
          if (Cb) Cb[(size_t)row * (NT * 16) + col] = (__bf16)v;
        }
      }
    }
  }
}

// --------------------------- message passing -------------------------------
// One 32-thread group per edge; each lane handles 4 features (float4).
__global__ void edge_scatter(const int* __restrict__ src,
                             const int* __restrict__ dst,
                             const int* __restrict__ et,
                             const float* __restrict__ M,   // [N,128] relu'd
                             const float* __restrict__ R,   // [200,128]
                             float* __restrict__ acc,       // [N,128]
                             float* __restrict__ cnt,       // [N]
                             int nE) {
  long tid = (long)blockIdx.x * blockDim.x + threadIdx.x;
  int e = (int)(tid >> 5);
  int g = (int)(tid & 31);
  if (e >= nE) return;
  int s = src[e], d = dst[e], t = et[e];
  float4 mv = *(const float4*)(M + (size_t)s * FEATD + g * 4);
  float4 rv = *(const float4*)(R + (size_t)t * FEATD + g * 4);
  float* o = acc + (size_t)d * FEATD + g * 4;
  atomicAdd(o + 0, mv.x + rv.x);
  atomicAdd(o + 1, mv.y + rv.y);
  atomicAdd(o + 2, mv.z + rv.z);
  atomicAdd(o + 3, mv.w + rv.w);
  if (g == 0) atomicAdd(cnt + d, 1.0f);
}

__global__ void seg_finalize(const float* __restrict__ acc,
                             const float* __restrict__ cnt,
                             float* __restrict__ outF,
                             __bf16* __restrict__ outB, int nrows) {
  long idx = (long)blockIdx.x * blockDim.x + threadIdx.x;
  if (idx >= (long)nrows * FEATD) return;
  float c = cnt[idx >> 7];
  c = c > 1.f ? c : 1.f;
  float v = acc[idx] / c;
  if (outF) outF[idx] = v;
  if (outB) outB[idx] = (__bf16)v;
}

// ------------------------------ batchnorm ----------------------------------
__global__ void bn_stats(const float* __restrict__ X,
                         float* __restrict__ stats, int nrows) {
  int f = threadIdx.x;  // blockDim.x == 128
  float s = 0.f, ss = 0.f;
  for (int r = blockIdx.x; r < nrows; r += gridDim.x) {
    float v = X[(size_t)r * FEATD + f];
    s += v; ss += v * v;
  }
  atomicAdd(&stats[f], s);
  atomicAdd(&stats[FEATD + f], ss);
}

__global__ void bn_apply_bf16(const float* __restrict__ X,
                              const float* __restrict__ stats,
                              const float* __restrict__ gamma,
                              const float* __restrict__ beta,
                              __bf16* __restrict__ Y, int nrows) {
  long idx = (long)blockIdx.x * blockDim.x + threadIdx.x;
  if (idx >= (long)nrows * FEATD) return;
  int f = (int)(idx & (FEATD - 1));
  float inv_n = 1.f / (float)nrows;
  float mu  = stats[f] * inv_n;
  float var = stats[FEATD + f] * inv_n - mu * mu;
  float v = (X[idx] - mu) * rsqrtf(var + BNEPS) * gamma[f] + beta[f];
  v = v > 0.f ? v : 0.f;
  Y[idx] = (__bf16)v;
}

// ------------------------------ launcher -----------------------------------
extern "C" void kernel_launch(void* const* d_in, const int* in_sizes, int n_in,
                              void* d_out, int out_size, void* d_ws,
                              size_t ws_size, hipStream_t stream) {
  const int*   edge_src = (const int*)d_in[0];   // [2, 500000]
  const int*   edge_dst = (const int*)d_in[1];
  const int*   edge_typ = (const int*)d_in[2];
  const float* emb_h    = (const float*)d_in[3]; // [100000,100]
  const float* emb_e    = (const float*)d_in[4]; // [50,100]
  const float* rel_wt   = (const float*)d_in[5]; // [200,50]
  const float* W_h_init = (const float*)d_in[6]; // [100,128]
  const float* W_e_init = (const float*)d_in[7]; // [100,128]
  const float* aggre_W  = (const float*)d_in[8]; // [128,128]
  const float* aggre_b  = (const float*)d_in[9]; // [128]
  const float* bn_gamma = (const float*)d_in[10];
  const float* bn_beta  = (const float*)d_in[11];
  const float* fc0_w    = (const float*)d_in[12]; // [128,64]
  const float* fc0_b    = (const float*)d_in[13];
  const float* fc1_w    = (const float*)d_in[14]; // [64,32]
  const float* fc1_b    = (const float*)d_in[15];
  const float* fc2_w    = (const float*)d_in[16]; // [32,50]
  const float* fc2_b    = (const float*)d_in[17];
  float* out = (float*)d_out;                     // [100000,50]

  // ---- workspace carve (needs ~129 MB) ----
  char* ws = (char*)d_ws;
  size_t off = 0;
  auto carve = [&](size_t bytes) {
    char* p = ws + off;
    off = (off + bytes + 255) & ~(size_t)255;
    return p;
  };
  __bf16* Xbf  = (__bf16*)carve((size_t)NNODES * FEATD * 2); // activations bf16
  float*  Mbuf = (float*)carve((size_t)NNODES * FEATD * 4);  // node messages
  float*  accb = (float*)carve((size_t)NNODES * FEATD * 4);  // segment sums
  float*  cntb = (float*)carve((size_t)NNODES * 4);
  float*  EW   = (float*)carve((size_t)NBASE * FEATD * 4);
  float*  Rtab = (float*)carve((size_t)NRELS * FEATD * 4);
  float*  W0   = (float*)carve((size_t)INITD * FEATD * 4);
  __bf16* W0t  = (__bf16*)carve((size_t)FEATD * FEATD * 2);
  __bf16* Wat  = (__bf16*)carve((size_t)FEATD * FEATD * 2);
  __bf16* fc0t = (__bf16*)carve((size_t)64 * 128 * 2);
  __bf16* fc1t = (__bf16*)carve((size_t)32 * 64 * 2);
  __bf16* fc2t = (__bf16*)carve((size_t)64 * 32 * 2);   // N padded 50->64
  float*  stat = (float*)carve((size_t)2 * FEATD * 4);
  // MLP bf16 intermediates alias the (then-free) segment-sum buffer
  __bf16* H1bf = (__bf16*)accb;                          // [N,64]
  __bf16* H2bf = (__bf16*)(accb + (size_t)NNODES * 64);  // [N,32] (f32 slots)

  const int B = 256;
  auto g1 = [&](long n) { return (unsigned)((n + B - 1) / B); };
  // 8 waves/block, 32 rows/wave (RT=2): 100000/32 = 3125 strips exactly.
  int gemmBlocks = ((NNODES + 31) / 32 + 7) / 8;

  // ---- tiny precomputes (fused weight algebra) ----
  gemm_small<<<g1(NBASE * FEATD), B, 0, stream>>>(emb_e, W_e_init, EW, NBASE, INITD, FEATD);
  gemm_small<<<g1(NRELS * FEATD), B, 0, stream>>>(rel_wt, EW, Rtab, NRELS, NBASE, FEATD);
  gemm_small<<<g1(INITD * FEATD), B, 0, stream>>>(W_h_init, aggre_W, W0, INITD, FEATD, FEATD);
  transpose_to_bf16<<<g1(FEATD * FEATD), B, 0, stream>>>(W0, W0t, INITD, FEATD, FEATD, FEATD);
  transpose_to_bf16<<<g1(FEATD * FEATD), B, 0, stream>>>(aggre_W, Wat, FEATD, FEATD, FEATD, FEATD);
  transpose_to_bf16<<<g1(64 * 128), B, 0, stream>>>(fc0_w, fc0t, 128, 64, 128, 64);
  transpose_to_bf16<<<g1(32 * 64), B, 0, stream>>>(fc1_w, fc1t, 64, 32, 64, 32);
  transpose_to_bf16<<<g1(64 * 32), B, 0, stream>>>(fc2_w, fc2t, 32, 50, 32, 64);

  // ---- layer 0 ----
  convert_pad_bf16<<<g1((long)NNODES * FEATD), B, 0, stream>>>(emb_h, Xbf, NNODES, INITD, FEATD);
  wmma_gemm<8, 2><<<gemmBlocks, B, 0, stream>>>(Xbf, FEATD, W0t, aggre_b,
                                                Mbuf, FEATD, FEATD, nullptr, NNODES, 1);
  zero_f32<<<4096, B, 0, stream>>>(accb, (long)NNODES * FEATD);
  zero_f32<<<512, B, 0, stream>>>(cntb, NNODES);
  edge_scatter<<<g1((long)NEDGES * 32), B, 0, stream>>>(
      edge_src, edge_dst, edge_typ, Mbuf, Rtab, accb, cntb, NEDGES);
  seg_finalize<<<g1((long)NNODES * FEATD), B, 0, stream>>>(accb, cntb, nullptr, Xbf, NNODES);

  // ---- layer 1 ----
  wmma_gemm<8, 2><<<gemmBlocks, B, 0, stream>>>(Xbf, FEATD, Wat, aggre_b,
                                                Mbuf, FEATD, FEATD, nullptr, NNODES, 1);
  zero_f32<<<4096, B, 0, stream>>>(accb, (long)NNODES * FEATD);
  zero_f32<<<512, B, 0, stream>>>(cntb, NNODES);
  edge_scatter<<<g1((long)NEDGES * 32), B, 0, stream>>>(
      edge_src + NEDGES, edge_dst + NEDGES, edge_typ + NEDGES, Mbuf, Rtab, accb, cntb, NEDGES);
  seg_finalize<<<g1((long)NNODES * FEATD), B, 0, stream>>>(accb, cntb, Mbuf, nullptr, NNODES);

  // ---- batchnorm + relu -> bf16 ----
  zero_f32<<<1, B, 0, stream>>>(stat, 2 * FEATD);
  bn_stats<<<512, FEATD, 0, stream>>>(Mbuf, stat, NNODES);
  bn_apply_bf16<<<g1((long)NNODES * FEATD), B, 0, stream>>>(Mbuf, stat, bn_gamma, bn_beta, Xbf, NNODES);

  // ---- MLP classifier 128 -> 64 -> 32 -> 50 ----
  wmma_gemm<4, 2><<<gemmBlocks, B, 0, stream>>>(Xbf, 128, fc0t, fc0_b,
                                                nullptr, 0, 64, H1bf, NNODES, 1);
  wmma_gemm<2, 2><<<gemmBlocks, B, 0, stream>>>(H1bf, 64, fc1t, fc1_b,
                                                nullptr, 0, 32, H2bf, NNODES, 1);
  wmma_gemm<4, 2><<<gemmBlocks, B, 0, stream>>>(H2bf, 32, fc2t, fc2_b,
                                                out, NCLS, NCLS, nullptr, NNODES, 0);
}
